// GCNNodeClassifier_48473000903026
// MI455X (gfx1250) — matrix operand order repaired and checked by reading, not compiled, not run on hardware
//
#include <hip/hip_runtime.h>

typedef __attribute__((ext_vector_type(16))) __bf16 v16bf;
typedef __attribute__((ext_vector_type(8)))  __bf16 v8bf;
typedef __attribute__((ext_vector_type(8)))  float  v8f;

#define NNODES 10000
#define NEDGES 320000
#define DIN    512
#define DHID   2048
#define NCLS   40
#define NPAD   48   // NCLS padded to 3 WMMA tiles

// ---------------- CSR build ----------------

__global__ void k_zero_i32(int* p, int n) {
    int i = blockIdx.x * blockDim.x + threadIdx.x;
    if (i < n) p[i] = 0;
}

__global__ void k_count_deg(const int* __restrict__ dst, int* __restrict__ counts, int e) {
    int i = blockIdx.x * blockDim.x + threadIdx.x;
    if (i < e) atomicAdd(&counts[dst[i]], 1);
}

__global__ void k_dinv(const int* __restrict__ counts, float* __restrict__ dinv, int n) {
    int i = blockIdx.x * blockDim.x + threadIdx.x;
    if (i < n) dinv[i] = rsqrtf((float)counts[i] + 1.0f);
}

// single-block exclusive scan of counts[0..n) -> rowptr[0..n]
__global__ void k_scan(const int* __restrict__ counts, int* __restrict__ rowptr, int n) {
    __shared__ int sums[1024];
    int t = threadIdx.x;
    int chunk = (n + 1023) / 1024;
    int base = t * chunk;
    int s = 0;
    for (int i = 0; i < chunk; ++i) {
        int idx = base + i;
        if (idx < n) s += counts[idx];
    }
    sums[t] = s;
    __syncthreads();
    for (int off = 1; off < 1024; off <<= 1) {
        int v = (t >= off) ? sums[t - off] : 0;
        __syncthreads();
        sums[t] += v;
        __syncthreads();
    }
    int excl = (t == 0) ? 0 : sums[t - 1];
    for (int i = 0; i < chunk; ++i) {
        int idx = base + i;
        if (idx < n) {
            rowptr[idx] = excl;
            excl += counts[idx];
        }
    }
    if (t == 1023) rowptr[n] = sums[1023];
}

__global__ void k_fill_csr(const int* __restrict__ src, const int* __restrict__ dst, int e,
                           const int* __restrict__ rowptr, int* __restrict__ fillc,
                           int* __restrict__ col) {
    int i = blockIdx.x * blockDim.x + threadIdx.x;
    if (i < e) {
        int d = dst[i];
        int pos = rowptr[d] + atomicAdd(&fillc[d], 1);
        col[pos] = src[i];
    }
}

// ---------------- weight transpose + bf16 convert ----------------

// W1 [512 x 2048] f32 -> W1t [2048 x 512] bf16 (N-major)
__global__ void k_w1t(const float* __restrict__ W1, __bf16* __restrict__ W1t) {
    int n = blockIdx.x;          // 0..2047
    int k = threadIdx.x;         // 0..511
    W1t[n * DIN + k] = (__bf16)W1[k * DHID + n];
}

// W2 [2048 x 40] f32 -> W2t [48 x 2048] bf16 (N-major, zero padded)
__global__ void k_w2t(const float* __restrict__ W2, __bf16* __restrict__ W2t) {
    int n = blockIdx.x;                          // 0..47
    int k = blockIdx.y * 256 + threadIdx.x;      // 0..2047
    W2t[n * DHID + k] = (n < NCLS) ? (__bf16)W2[k * NCLS + n] : (__bf16)0.0f;
}

// ---------------- propagate x (f32 gather), write bf16 A ----------------

__global__ void k_agg_x(const float* __restrict__ x, const int* __restrict__ rowptr,
                        const int* __restrict__ col, const float* __restrict__ dinv,
                        __bf16* __restrict__ aggbf) {
    int node = blockIdx.x;
    int f = threadIdx.x;             // features f and f+256
    float di = dinv[node];
    float a0 = 0.0f, a1 = 0.0f;
    int s = rowptr[node], e = rowptr[node + 1];
    for (int p = s; p < e; ++p) {
        int j = col[p];
        float c = dinv[j] * di;
        a0 += c * x[j * DIN + f];
        a1 += c * x[j * DIN + f + 256];
    }
    float cs = di * di;
    a0 += cs * x[node * DIN + f];
    a1 += cs * x[node * DIN + f + 256];
    aggbf[node * DIN + f]       = (__bf16)a0;
    aggbf[node * DIN + f + 256] = (__bf16)a1;
}

// ---------------- WMMA fragment helpers ----------------

// A-matrix 16x32 bf16 per ISA layout: lane m = lane&15, half = lane>>4
// elements i = K 16*half + i (contiguous, matching B)
__device__ inline v16bf load_a_frag(const __bf16* __restrict__ arow, int k0, int half) {
    v8bf lo = *(const v8bf*)(arow + k0 + 16 * half);
    v8bf hi = *(const v8bf*)(arow + k0 + 16 * half + 8);
    return __builtin_shufflevector(lo, hi, 0, 1, 2, 3, 4, 5, 6, 7,
                                           8, 9, 10, 11, 12, 13, 14, 15);
}

// B-matrix 32x16 bf16, N-major storage: lane n = lane&15, elements i = K 16*half + i
__device__ inline v16bf load_b_frag(const __bf16* __restrict__ brow, int k0, int half) {
    return *(const v16bf*)(brow + k0 + 16 * half);
}

// ---------------- GEMM1: agg[10000x512] @ W1t -> relu(+b1) -> h bf16 [10000x2048] ----------------
// Each wave computes a 32(M) x 64(N) tile: 2 A frags x 4 B frags = 8 WMMAs per K-step.

__global__ __launch_bounds__(256)
void k_gemm1(const __bf16* __restrict__ A, const __bf16* __restrict__ Bt,
             const float* __restrict__ b1, __bf16* __restrict__ H) {
    int wave = threadIdx.x >> 5;                 // 0..7
    int lane = threadIdx.x & 31;
    int half = lane >> 4;
    int l16  = lane & 15;
    int m0 = blockIdx.x * 32;                              // 313 m-tiles of 32 rows
    int n0 = (blockIdx.y * 8 + wave) * 64;                 // wave covers N 64 wide

    int ra0 = m0 + l16;                                    // A row for frag 0
    int ra1 = m0 + 16 + l16;                               // A row for frag 1 (clamp OOB)
    if (ra1 >= NNODES) ra1 = NNODES - 1;                   // keep loads in-bounds
    const __bf16* arow0 = A + (size_t)ra0 * DIN;
    const __bf16* arow1 = A + (size_t)ra1 * DIN;

    v8f acc[2][4] = {};
    for (int k0 = 0; k0 < DIN; k0 += 32) {
        v16bf a0 = load_a_frag(arow0, k0, half);
        v16bf a1 = load_a_frag(arow1, k0, half);
#pragma unroll
        for (int t = 0; t < 4; ++t) {
            const __bf16* brow = Bt + (size_t)(n0 + t * 16 + l16) * DIN;
            v16bf b = load_b_frag(brow, k0, half);
            acc[0][t] = __builtin_amdgcn_wmma_f32_16x16x32_bf16(
                false, a0, false, b, (short)0, acc[0][t], false, false);
            acc[1][t] = __builtin_amdgcn_wmma_f32_16x16x32_bf16(
                false, a1, false, b, (short)0, acc[1][t], false, false);
        }
    }
#pragma unroll
    for (int t = 0; t < 4; ++t) {
        int n = n0 + t * 16 + l16;
        float bias = b1[n];
#pragma unroll
        for (int g = 0; g < 2; ++g) {
#pragma unroll
            for (int r = 0; r < 8; ++r) {
                int m = m0 + g * 16 + r + 8 * half;
                if (m < NNODES) {
                    float v = acc[g][t][r] + bias;
                    H[(size_t)m * DHID + n] = (__bf16)(v > 0.0f ? v : 0.0f);
                }
            }
        }
    }
}

// ---------------- GEMM2: h[10000x2048] @ W2t -> T f32 [10000x40] ----------------

__global__ __launch_bounds__(256)
void k_gemm2(const __bf16* __restrict__ H, const __bf16* __restrict__ Bt,
             float* __restrict__ T) {
    int wave = threadIdx.x >> 5;
    int lane = threadIdx.x & 31;
    int half = lane >> 4;
    int l16  = lane & 15;
    int mtile = blockIdx.x * 8 + wave;
    if (mtile >= NNODES / 16) return;
    int m0 = mtile * 16;
    const __bf16* arow = H + (size_t)(m0 + l16) * DHID;

    v8f acc[3] = {};
    for (int k0 = 0; k0 < DHID; k0 += 32) {
        v16bf a = load_a_frag(arow, k0, half);
#pragma unroll
        for (int t = 0; t < 3; ++t) {
            const __bf16* brow = Bt + (size_t)(t * 16 + l16) * DHID;
            v16bf b = load_b_frag(brow, k0, half);
            acc[t] = __builtin_amdgcn_wmma_f32_16x16x32_bf16(
                false, a, false, b, (short)0, acc[t], false, false);
        }
    }
#pragma unroll
    for (int t = 0; t < 3; ++t) {
        int n = t * 16 + l16;
        if (n < NCLS) {
#pragma unroll
            for (int r = 0; r < 8; ++r) {
                int m = m0 + r + 8 * half;
                T[(size_t)m * NCLS + n] = acc[t][r];
            }
        }
    }
}

// ---------------- propagate T + b2 -> out ----------------

__global__ void k_agg_out(const float* __restrict__ T, const int* __restrict__ rowptr,
                          const int* __restrict__ col, const float* __restrict__ dinv,
                          const float* __restrict__ b2, float* __restrict__ out) {
    int node = blockIdx.x * 4 + (threadIdx.x >> 5);
    if (node >= NNODES) return;
    int lane = threadIdx.x & 31;       // lanes cover features lane and lane+32
    float di = dinv[node];
    float a0 = 0.0f, a1 = 0.0f;
    int s = rowptr[node], e = rowptr[node + 1];
    for (int p = s; p < e; ++p) {
        int j = col[p];
        float c = dinv[j] * di;
        a0 += c * T[j * NCLS + lane];
        if (lane < NCLS - 32) a1 += c * T[j * NCLS + 32 + lane];
    }
    float cs = di * di;
    a0 += cs * T[node * NCLS + lane];
    out[node * NCLS + lane] = a0 + b2[lane];
    if (lane < NCLS - 32) {
        a1 += cs * T[node * NCLS + 32 + lane];
        out[node * NCLS + 32 + lane] = a1 + b2[32 + lane];
    }
}

// ---------------- launcher ----------------

extern "C" void kernel_launch(void* const* d_in, const int* in_sizes, int n_in,
                              void* d_out, int out_size, void* d_ws, size_t ws_size,
                              hipStream_t stream) {
    const float* x   = (const float*)d_in[0];
    const int*   ei  = (const int*)d_in[1];    // [2, E]: src row then dst row
    const float* W1  = (const float*)d_in[2];
    const float* b1  = (const float*)d_in[3];
    const float* W2  = (const float*)d_in[4];
    const float* b2  = (const float*)d_in[5];
    float* out = (float*)d_out;

    const int* src = ei;
    const int* dst = ei + NEDGES;

    // carve workspace (256B aligned chunks)
    char* p = (char*)d_ws;
    auto carve = [&](size_t bytes) -> void* {
        void* r = (void*)p;
        p += (bytes + 255) & ~(size_t)255;
        return r;
    };
    int*    counts = (int*)carve(NNODES * sizeof(int));
    int*    rowptr = (int*)carve((NNODES + 1) * sizeof(int));
    int*    fillc  = (int*)carve(NNODES * sizeof(int));
    float*  dinv   = (float*)carve(NNODES * sizeof(float));
    int*    col    = (int*)carve(NEDGES * sizeof(int));
    __bf16* aggbf  = (__bf16*)carve((size_t)NNODES * DIN * sizeof(__bf16));
    __bf16* W1t    = (__bf16*)carve((size_t)DHID * DIN * sizeof(__bf16));
    __bf16* hbf    = (__bf16*)carve((size_t)NNODES * DHID * sizeof(__bf16));
    __bf16* W2t    = (__bf16*)carve((size_t)NPAD * DHID * sizeof(__bf16));
    float*  T      = (float*)carve((size_t)NNODES * NCLS * sizeof(float));

    // 1) degree + dinv
    k_zero_i32<<<(NNODES + 255) / 256, 256, 0, stream>>>(counts, NNODES);
    k_zero_i32<<<(NNODES + 255) / 256, 256, 0, stream>>>(fillc, NNODES);
    k_count_deg<<<(NEDGES + 255) / 256, 256, 0, stream>>>(dst, counts, NEDGES);
    k_dinv<<<(NNODES + 255) / 256, 256, 0, stream>>>(counts, dinv, NNODES);

    // 2) CSR
    k_scan<<<1, 1024, 0, stream>>>(counts, rowptr, NNODES);
    k_fill_csr<<<(NEDGES + 255) / 256, 256, 0, stream>>>(src, dst, NEDGES, rowptr, fillc, col);

    // 3) weights -> bf16 transposed
    k_w1t<<<DHID, DIN, 0, stream>>>(W1, W1t);
    k_w2t<<<dim3(NPAD, DHID / 256), 256, 0, stream>>>(W2, W2t);

    // 4) propagate x -> bf16 A
    k_agg_x<<<NNODES, 256, 0, stream>>>(x, rowptr, col, dinv, aggbf);

    // 5) GEMM1 + bias + relu -> h bf16  (32-row M tiles: 313 covers 10000 with clamping)
    k_gemm1<<<dim3((NNODES + 31) / 32, DHID / (8 * 64)), 256, 0, stream>>>(aggbf, W1t, b1, hbf);

    // 6) GEMM2 -> T f32
    k_gemm2<<<(NNODES / 16 + 7) / 8, 256, 0, stream>>>(hbf, W2t, T);

    // 7) propagate T + b2 -> out
    k_agg_out<<<(NNODES + 3) / 4, 128, 0, stream>>>(T, rowptr, col, dinv, b2, out);
}
